// DualStoSelfAttention_77146202571311
// MI455X (gfx1250) — compile-verified, matrix operand order checked by reading, not run on hardware
//
#include <hip/hip_runtime.h>
#include <hip/hip_bf16.h>

// ---- problem constants ----
#define BB   4
#define SS   1024
#define HSZ  1024
#define NH   16
#define HD   64
#define NC   256
#define MTOT (BB*SS)     // 4096 rows

typedef __attribute__((ext_vector_type(4)))  __bf16 bf16x4;
typedef __attribute__((ext_vector_type(8)))  __bf16 bf16x8;
typedef __attribute__((ext_vector_type(16))) __bf16 bf16x16;
typedef __attribute__((ext_vector_type(8)))  float  f32x8;
typedef __attribute__((ext_vector_type(4)))  unsigned int u32x4;
typedef __attribute__((ext_vector_type(4)))  int i32x4;
typedef __attribute__((ext_vector_type(8)))  int i32x8;

union Frag { bf16x16 v; bf16x8 h[2]; };

__device__ __forceinline__ unsigned int lds_off(const void* p) {
    // generic LDS pointer: addr[31:0] is the LDS byte offset (flat aperture rule)
    return (unsigned int)(unsigned long long)p;
}
__device__ __forceinline__ void async_load_lds_b128(unsigned int ldsaddr, const void* gaddr) {
    asm volatile("global_load_async_to_lds_b128 %0, %1, off"
                 :: "v"(ldsaddr), "v"((unsigned long long)gaddr) : "memory");
}
__device__ __forceinline__ void async_load_lds_b64(unsigned int ldsaddr, const void* gaddr) {
    asm volatile("global_load_async_to_lds_b64 %0, %1, off"
                 :: "v"(ldsaddr), "v"((unsigned long long)gaddr) : "memory");
}
__device__ __forceinline__ void wait_async0() {
    asm volatile("s_wait_asynccnt 0x0" ::: "memory");
}

// ---------------- fp32 -> bf16 convert (vectorized x4) ----------------
__global__ void cvt4_kernel(const float* __restrict__ src, __bf16* __restrict__ dst, int n4) {
    for (int i = blockIdx.x * blockDim.x + threadIdx.x; i < n4; i += gridDim.x * blockDim.x) {
        float4 f = ((const float4*)src)[i];
        bf16x4 o;
        o[0] = (__bf16)f.x; o[1] = (__bf16)f.y; o[2] = (__bf16)f.z; o[3] = (__bf16)f.w;
        ((bf16x4*)dst)[i] = o;
    }
}

// ---------------- generic bf16 GEMM:  C[M,N] = A[M,K] * W[N,K]^T + bias ----------------
// Block tile 128(M) x 64(N): 8 waves stacked in M share one 64-row weight tile,
// staged to LDS via async loads, double-buffered (DMA of tile t+1 overlaps WMMA of t).
// MODE 0: fp32 out row-major [M,N]; MODE 1: bf16 out -> [B,NH,S,HD]; MODE 2: fp32 -> [B,NH,S,HD].
template<int MODE>
__global__ void gemm_bf16_kernel(const __bf16* __restrict__ A,
                                 const __bf16* __restrict__ W,
                                 const float*  __restrict__ bias,
                                 float*  __restrict__ Cf,
                                 __bf16* __restrict__ Cb)
{
    const int K = HSZ, N = HSZ;
    __shared__ __bf16 sB[2][64 * 32];                     // 2 x 4KB weight tiles

    int tid  = threadIdx.x;
    int wblk = tid >> 5;
    int lane = tid & 31;
    int lm   = lane & 15;
    int sel  = lane >> 4;
    int m0 = (blockIdx.x >> 4) * 128 + wblk * 16;         // 32 M-blocks
    int n0 = (blockIdx.x & 15) * 64;                      // 16 N-blocks

    // async staging slice: thread copies 16B; row = tid/4, chunk = (tid%4)*8 elems
    int srow = tid >> 2, schunk = (tid & 3) * 8;
    const __bf16* wsrc = W + (size_t)(n0 + srow) * K + schunk;
    unsigned int dst0 = lds_off(&sB[0][srow * 32 + schunk]);
    unsigned int dst1 = lds_off(&sB[1][srow * 32 + schunk]);

    const __bf16* arow = A + (size_t)(m0 + lm) * K + sel * 8;

    f32x8 acc0 = {}, acc1 = {}, acc2 = {}, acc3 = {};

    async_load_lds_b128(dst0, wsrc);                      // prefetch tile 0
    for (int t = 0; t < K / 32; ++t) {
        int k0 = t * 32;
        wait_async0();                                    // own slice of tile t landed
        __syncthreads();                                  // all waves' slices landed
        if (t + 1 < K / 32)                               // overlap DMA of t+1 with compute of t
            async_load_lds_b128((t & 1) ? dst0 : dst1, wsrc + k0 + 32);

        __builtin_prefetch(arow + k0 + 256, 0, 1);        // global_prefetch_b8 (A stream)
        const __bf16* bb = &sB[t & 1][0];
        Frag a, b0, b1, b2, b3;
        a.h[0]  = *(const bf16x8*)(arow + k0);
        a.h[1]  = *(const bf16x8*)(arow + k0 + 16);
        b0.h[0] = *(const bf16x8*)(bb + ( 0 + lm) * 32 + sel * 8);
        b0.h[1] = *(const bf16x8*)(bb + ( 0 + lm) * 32 + sel * 8 + 16);
        b1.h[0] = *(const bf16x8*)(bb + (16 + lm) * 32 + sel * 8);
        b1.h[1] = *(const bf16x8*)(bb + (16 + lm) * 32 + sel * 8 + 16);
        b2.h[0] = *(const bf16x8*)(bb + (32 + lm) * 32 + sel * 8);
        b2.h[1] = *(const bf16x8*)(bb + (32 + lm) * 32 + sel * 8 + 16);
        b3.h[0] = *(const bf16x8*)(bb + (48 + lm) * 32 + sel * 8);
        b3.h[1] = *(const bf16x8*)(bb + (48 + lm) * 32 + sel * 8 + 16);
        acc0 = __builtin_amdgcn_wmma_f32_16x16x32_bf16(false, a.v, false, b0.v, (short)0, acc0, false, false);
        acc1 = __builtin_amdgcn_wmma_f32_16x16x32_bf16(false, a.v, false, b1.v, (short)0, acc1, false, false);
        acc2 = __builtin_amdgcn_wmma_f32_16x16x32_bf16(false, a.v, false, b2.v, (short)0, acc2, false, false);
        acc3 = __builtin_amdgcn_wmma_f32_16x16x32_bf16(false, a.v, false, b3.v, (short)0, acc3, false, false);
    }

    f32x8 accs[4] = {acc0, acc1, acc2, acc3};
    #pragma unroll
    for (int j = 0; j < 4; ++j) {
        int n = n0 + j * 16 + lm;
        float bvs = bias[n];
        #pragma unroll
        for (int r = 0; r < 8; ++r) {
            int m = m0 + r + sel * 8;                     // C layout: lanes0-15 M=r, lanes16-31 M=r+8
            float val = accs[j][r] + bvs;
            if (MODE == 0) {
                Cf[(size_t)m * N + n] = val;
            } else {
                int b = m >> 10, s = m & 1023;
                int h = n >> 6,  d = n & 63;
                size_t idx = ((size_t)((b * NH + h) << 10) + s) * HD + d;
                if (MODE == 1) Cb[idx] = (__bf16)val;
                else           Cf[idx] = val;
            }
        }
    }
}

// ---------------- VQ: logits = kk . centroid, +g1, hard argmax, gather centroid col ----------------
// Centroid table (64x256 fp32, 64KB) is DMAed into LDS by the Tensor Data Mover.
__global__ void vq_kernel(const float* __restrict__ kk,    // [MTOT, HSZ] fp32 (K projection)
                          const float* __restrict__ cent,  // [HD, NC]
                          const float* __restrict__ g1,    // [B,S,NH,NC]
                          __bf16* __restrict__ kq)         // [B,NH,S,HD] bf16
{
    __shared__ float sc[HD * NC];                          // 64 KB of 320 KB LDS
    if (threadIdx.x < 32) {                                // one wave issues the TDM descriptor
        unsigned long long ga = (unsigned long long)cent;
        u32x4 g0 = { 1u,                                   // count=1 (valid user D#)
                     lds_off(&sc[0]),                      // lds_addr  [63:32]
                     (unsigned int)ga,                     // global_addr lo [95:64]
                     ((unsigned int)(ga >> 32) & 0x01FFFFFFu) | (2u << 30) }; // addr hi | type=2
        i32x8 g1d = { (int)(2u << 16),                     // data_size=4B
                      (int)(NC << 16),                     // tensor_dim0[15:0] @ bits 63:48
                      (int)(HD << 16),                     // tensor_dim1[15:0] @ bits 95:80 (dim0 hi16=0)
                      (int)(NC << 16),                     // tile_dim0 @ bits 127:112
                      (int)HD,                             // tile_dim1 (tile_dim2=0)
                      (int)NC,                             // tensor_dim0_stride lo32
                      0, 0 };
        i32x4 gz = {0, 0, 0, 0};
#if __clang_major__ >= 23
        i32x8 gz8 = {0, 0, 0, 0, 0, 0, 0, 0};
        __builtin_amdgcn_tensor_load_to_lds(g0, g1d, gz, gz, gz8, 0);
#else
        __builtin_amdgcn_tensor_load_to_lds(g0, g1d, gz, gz, 0);
#endif
        __builtin_amdgcn_s_wait_tensorcnt(0);
    }
    __syncthreads();

    int item = blockIdx.x * blockDim.x + threadIdx.x;      // (b*S+s)*NH + h, 65536 items
    int h   = item & (NH - 1);
    int row = item >> 4;                                   // b*S + s
    const float4* kr4 = (const float4*)(kk + (size_t)row * HSZ + h * HD);
    float kv[HD];
    #pragma unroll
    for (int d4 = 0; d4 < HD / 4; ++d4) {
        float4 f = kr4[d4];
        kv[d4*4+0] = f.x; kv[d4*4+1] = f.y; kv[d4*4+2] = f.z; kv[d4*4+3] = f.w;
    }
    const float* g = g1 + (size_t)item * NC;
    float best = -3.4e38f; int bc = 0;
    for (int c = 0; c < NC; c += 4) {
        float d0 = 0.f, d1 = 0.f, d2 = 0.f, d3 = 0.f;
        #pragma unroll
        for (int d = 0; d < HD; ++d) {
            float k = kv[d];
            const float* cp = sc + d * NC + c;             // uniform across wave -> LDS broadcast
            d0 += k * cp[0]; d1 += k * cp[1]; d2 += k * cp[2]; d3 += k * cp[3];
        }
        float4 gg = *(const float4*)(g + c);
        d0 += gg.x; d1 += gg.y; d2 += gg.z; d3 += gg.w;
        if (d0 > best) { best = d0; bc = c + 0; }          // strict > keeps first max (jnp.argmax)
        if (d1 > best) { best = d1; bc = c + 1; }
        if (d2 > best) { best = d2; bc = c + 2; }
        if (d3 > best) { best = d3; bc = c + 3; }
    }
    int b = row >> 10, s = row & 1023;
    __bf16* out = kq + ((size_t)((b * NH + h) << 10) + s) * HD;
    #pragma unroll
    for (int d = 0; d < HD; d += 4) {
        bf16x4 o;
        o[0] = (__bf16)sc[(d+0) * NC + bc];
        o[1] = (__bf16)sc[(d+1) * NC + bc];
        o[2] = (__bf16)sc[(d+2) * NC + bc];
        o[3] = (__bf16)sc[(d+3) * NC + bc];
        *(bf16x4*)(out + d) = o;
    }
}

// ---------------- fused QK^T + g2 + running argmax over keys ----------------
// 8 waves per block share one (b,h): each wave owns a 16-query strip; the shared
// 16x64 key tile is async-staged to LDS, double-buffered. Keys are WMMA M-dim,
// queries N-dim so each lane owns one query; per-query keys split lane vs lane^16.
__global__ void attn_kernel(const __bf16* __restrict__ qb,   // [B,NH,S,HD]
                            const __bf16* __restrict__ kqb,  // [B,NH,S,HD]
                            const float*  __restrict__ g2,   // [B,NH,S,S]
                            int* __restrict__ attn_idx)      // [B,NH,S]
{
    __shared__ __bf16 sK[2][16 * HD];                        // 2 x 2KB key tiles

    int tid  = threadIdx.x;
    int wblk = tid >> 5;
    int lane = tid & 31;
    int lm   = lane & 15;
    int sel  = lane >> 4;
    int bh = blockIdx.x >> 3;                                // 8 blocks per (b,h)
    int q0 = ((blockIdx.x & 7) * 8 + wblk) << 4;             // 16-query strip per wave

    const __bf16* qrow = qb + ((size_t)((bh << 10) + q0 + lm)) * HD + sel * 8;
    Frag bq0, bq1;                                           // loop-invariant query fragments
    bq0.h[0] = *(const bf16x8*)(qrow);      bq0.h[1] = *(const bf16x8*)(qrow + 16);
    bq1.h[0] = *(const bf16x8*)(qrow + 32); bq1.h[1] = *(const bf16x8*)(qrow + 48);

    // async staging slice: thread copies 8B; row = tid/16, chunk = (tid%16)*4 elems
    const __bf16* kbase = kqb + (size_t)(bh << 10) * HD + (size_t)(tid >> 4) * HD + (tid & 15) * 4;
    unsigned int kd0 = lds_off(&sK[0][(tid >> 4) * HD + (tid & 15) * 4]);
    unsigned int kd1 = lds_off(&sK[1][(tid >> 4) * HD + (tid & 15) * 4]);

    const float* gbase = g2 + (size_t)((bh << 10) + q0 + lm) * SS + sel * 8;

    float best = -3.4e38f; int bidx = 0;
    async_load_lds_b64(kd0, kbase);                          // prefetch key tile 0
    for (int t = 0; t < SS / 16; ++t) {
        int kb = t * 16;
        wait_async0();
        __syncthreads();
        if (t + 1 < SS / 16)
            async_load_lds_b64((t & 1) ? kd0 : kd1, kbase + (size_t)(kb + 16) * HD);

        const __bf16* kt = &sK[t & 1][0];
        Frag a0, a1;
        a0.h[0] = *(const bf16x8*)(kt + lm * HD + sel * 8);
        a0.h[1] = *(const bf16x8*)(kt + lm * HD + sel * 8 + 16);
        a1.h[0] = *(const bf16x8*)(kt + lm * HD + sel * 8 + 32);
        a1.h[1] = *(const bf16x8*)(kt + lm * HD + sel * 8 + 48);
        f32x8 acc = {};
        acc = __builtin_amdgcn_wmma_f32_16x16x32_bf16(false, a0.v, false, bq0.v, (short)0, acc, false, false);
        acc = __builtin_amdgcn_wmma_f32_16x16x32_bf16(false, a1.v, false, bq1.v, (short)0, acc, false, false);
        float4 ga = *(const float4*)(gbase + kb);
        float4 gc = *(const float4*)(gbase + kb + 4);
        float gv[8] = {ga.x, ga.y, ga.z, ga.w, gc.x, gc.y, gc.z, gc.w};
        int keyb = kb + sel * 8;
        #pragma unroll
        for (int r = 0; r < 8; ++r) {
            float v = acc[r] + gv[r];
            if (v > best) { best = v; bidx = keyb + r; }
        }
    }
    // each query's 16 keys/tile are split between lane and lane^16: combine
    float ob = __shfl_xor(best, 16, 32);
    int   oi = __shfl_xor(bidx, 16, 32);
    if (ob > best || (ob == best && oi < bidx)) { best = ob; bidx = oi; }
    if (lane < 16) attn_idx[(bh << 10) + q0 + lane] = bidx;
}

// ---------------- one-hot attention == gather of V rows (+ dual mask) ----------------
__global__ void gather_kernel(const float* __restrict__ vf,    // [B,NH,S,HD] fp32
                              const int*   __restrict__ aidx,  // [B,NH,S]
                              const float* __restrict__ mask,  // [B,S]
                              __bf16* __restrict__ outbf)      // [MTOT, HSZ] bf16
{
    int e = blockIdx.x * blockDim.x + threadIdx.x;             // 4M elements
    int d = e & 63;
    int h = (e >> 6) & 15;
    int s = (e >> 10) & 1023;
    int b = e >> 20;
    int bh = b * NH + h;
    int ki = aidx[(bh << 10) + s];
    float sc = mask[(b << 10) + s] * mask[(b << 10) + ki];
    float val = vf[((size_t)(bh << 10) + ki) * HD + d] * sc;
    outbf[e] = (__bf16)val;
}

// ---------------- residual + layernorm ----------------
__global__ void ln_kernel(const float* __restrict__ proj, const float* __restrict__ inp,
                          const float* __restrict__ gamma, const float* __restrict__ beta,
                          float* __restrict__ out)
{
    int row = blockIdx.x;                 // 4096 rows
    int tid = threadIdx.x;                // 256 threads
    const float* pr = proj + (size_t)row * HSZ;
    const float* ir = inp  + (size_t)row * HSZ;
    float x[4]; float s = 0.f;
    #pragma unroll
    for (int i = 0; i < 4; ++i) { int c = tid + i * 256; x[i] = pr[c] + ir[c]; s += x[i]; }

    __shared__ float red[8];
    for (int o = 16; o > 0; o >>= 1) s += __shfl_xor(s, o, 32);
    int wid = tid >> 5, ln = tid & 31;
    if (ln == 0) red[wid] = s;
    __syncthreads();
    float tot = 0.f;
    #pragma unroll
    for (int w = 0; w < 8; ++w) tot += red[w];
    float mu = tot * (1.0f / HSZ);
    __syncthreads();

    float vs = 0.f;
    #pragma unroll
    for (int i = 0; i < 4; ++i) { float dlt = x[i] - mu; vs += dlt * dlt; }
    for (int o = 16; o > 0; o >>= 1) vs += __shfl_xor(vs, o, 32);
    if (ln == 0) red[wid] = vs;
    __syncthreads();
    float vt = 0.f;
    #pragma unroll
    for (int w = 0; w < 8; ++w) vt += red[w];
    float rs = rsqrtf(vt * (1.0f / HSZ) + 1e-5f);
    #pragma unroll
    for (int i = 0; i < 4; ++i) {
        int c = tid + i * 256;
        out[(size_t)row * HSZ + c] = (x[i] - mu) * rs * gamma[c] + beta[c];
    }
}

extern "C" void kernel_launch(void* const* d_in, const int* in_sizes, int n_in,
                              void* d_out, int out_size, void* d_ws, size_t ws_size,
                              hipStream_t stream)
{
    const float* input = (const float*)d_in[0];
    const float* mask  = (const float*)d_in[1];
    const float* Wq = (const float*)d_in[2];  const float* bq = (const float*)d_in[3];
    const float* Wk = (const float*)d_in[4];  const float* bk = (const float*)d_in[5];
    const float* Wv = (const float*)d_in[6];  const float* bv = (const float*)d_in[7];
    const float* Wo = (const float*)d_in[8];  const float* bo = (const float*)d_in[9];
    const float* gamma = (const float*)d_in[10];
    const float* beta  = (const float*)d_in[11];
    const float* cent  = (const float*)d_in[12];
    const float* g1 = (const float*)d_in[13];
    const float* g2 = (const float*)d_in[14];
    float* out = (float*)d_out;

    char* ws = (char*)d_ws;
    const size_t MB = 1024u * 1024u;
    __bf16* inp_bf = (__bf16*)(ws + 0);        // 8MB (reused as bf16 attention output)
    __bf16* Wq_bf  = (__bf16*)(ws +  8 * MB);  // 2MB each
    __bf16* Wk_bf  = (__bf16*)(ws + 10 * MB);
    __bf16* Wv_bf  = (__bf16*)(ws + 12 * MB);
    __bf16* Wo_bf  = (__bf16*)(ws + 14 * MB);
    __bf16* q_bf   = (__bf16*)(ws + 16 * MB);  // 8MB [B,NH,S,HD]
    __bf16* kq_bf  = (__bf16*)(ws + 24 * MB);  // 8MB [B,NH,S,HD]
    float*  kk_f   = (float*)(ws + 32 * MB);   // 16MB (reused as O-proj output)
    float*  v_f    = (float*)(ws + 48 * MB);   // 16MB [B,NH,S,HD]
    int*    aidx   = (int*)(ws + 64 * MB);     // 256KB

    // bf16 conversions
    cvt4_kernel<<<2048, 256, 0, stream>>>(input, inp_bf, MTOT * HSZ / 4);
    cvt4_kernel<<<1024, 256, 0, stream>>>(Wq, Wq_bf, HSZ * HSZ / 4);
    cvt4_kernel<<<1024, 256, 0, stream>>>(Wk, Wk_bf, HSZ * HSZ / 4);
    cvt4_kernel<<<1024, 256, 0, stream>>>(Wv, Wv_bf, HSZ * HSZ / 4);
    cvt4_kernel<<<1024, 256, 0, stream>>>(Wo, Wo_bf, HSZ * HSZ / 4);

    // projections (512 blocks x 8 waves; 128x64 block tiles, async-staged weights)
    gemm_bf16_kernel<1><<<512, 256, 0, stream>>>(inp_bf, Wq_bf, bq, nullptr, q_bf);
    gemm_bf16_kernel<0><<<512, 256, 0, stream>>>(inp_bf, Wk_bf, bk, kk_f, nullptr);
    gemm_bf16_kernel<2><<<512, 256, 0, stream>>>(inp_bf, Wv_bf, bv, v_f, nullptr);

    // VQ hard gumbel argmax + centroid gather (TDM-staged centroid table)
    vq_kernel<<<256, 256, 0, stream>>>(kk_f, cent, g1, kq_bf);

    // fused QK^T + noise + hard argmax (async-staged key tiles)
    attn_kernel<<<512, 256, 0, stream>>>(q_bf, kq_bf, g2, aidx);

    // one-hot attention -> V gather (writes bf16, reuses inp_bf region)
    gather_kernel<<<MTOT * HSZ / 256, 256, 0, stream>>>(v_f, aidx, mask, inp_bf);

    // output projection (reuses kk_f region)
    float* proj = kk_f;
    gemm_bf16_kernel<0><<<512, 256, 0, stream>>>(inp_bf, Wo_bf, bo, proj, nullptr);

    // residual + layernorm
    ln_kernel<<<MTOT, 256, 0, stream>>>(proj, input, gamma, beta, out);
}
